// Subgraph2DGNNEncoder_88416196756193
// MI455X (gfx1250) — matrix operand order repaired and checked by reading, not compiled, not run on hardware
//
#include <hip/hip_runtime.h>
#include <hip/hip_bf16.h>

#define H    128
#define HT   8          // H / 16 column tiles
#define KST  4          // H / 32 k-steps
#define ATOM_V 120
#define BOND_V 5

typedef __attribute__((ext_vector_type(16))) _Float16 v16h;
typedef __attribute__((ext_vector_type(8)))  _Float16 v8h;
typedef __attribute__((ext_vector_type(8)))  float    v8f;
typedef unsigned int u32x4 __attribute__((ext_vector_type(4)));
typedef int          i32x8 __attribute__((ext_vector_type(8)));
typedef int          i32x4 __attribute__((ext_vector_type(4)));

// ---------------------------------------------------------------------------
// Pack one [H][H] row-major (k-major) f32 weight matrix into WMMA B-fragment
// order: packed[((kt*HT + nt)*32 + lane)*16 + idx] = w[k][n]
//   k = kt*32 + 8*(lane>>4) + (idx&7) + 16*(idx>>3),  n = nt*16 + (lane&15)
// so each lane's 16 f16 (32B) fragment chunk is contiguous in memory.
// ---------------------------------------------------------------------------
__global__ void pack_weight_kernel(const float* __restrict__ w,
                                   _Float16* __restrict__ p) {
  int t = blockIdx.x * blockDim.x + threadIdx.x;      // 0 .. 16383
  if (t >= H * H) return;
  int idx  = t & 15;
  int lane = (t >> 4) & 31;
  int nt   = (t >> 9) & 7;
  int kt   = (t >> 12) & 3;
  int k = kt * 32 + ((lane >> 4) << 3) + (idx & 7) + ((idx >> 3) << 4);
  int n = nt * 16 + (lane & 15);
  p[t] = (_Float16)w[k * H + n];
}

// ---------------------------------------------------------------------------
// Node embedding gather: x[i,:] = atom_emb[clip(type[i]),:]
// ---------------------------------------------------------------------------
__global__ void embed_nodes_kernel(const int* __restrict__ atom_types,
                                   const float* __restrict__ atom_emb,
                                   float* __restrict__ x, int N) {
  int gt   = blockIdx.x * blockDim.x + threadIdx.x;
  int node = gt >> 5;
  int lane = gt & 31;
  if (node >= N) return;
  int at = atom_types[node];
  at = at < 0 ? 0 : (at > ATOM_V - 1 ? ATOM_V - 1 : at);
  *(float4*)(x + (size_t)node * H + lane * 4) =
      *(const float4*)(atom_emb + (size_t)at * H + lane * 4);
}

// ---------------------------------------------------------------------------
// Edge messages: agg[dst] += relu(x[src] + bond_emb[etype])
// one wave per edge, float4 per lane, f32 global atomics
// ---------------------------------------------------------------------------
__global__ void edge_message_kernel(const float* __restrict__ x,
                                    const float* __restrict__ bond_emb,
                                    const int* __restrict__ src,
                                    const int* __restrict__ dst,
                                    const int* __restrict__ etype,
                                    float* __restrict__ agg, int E) {
  int gt   = blockIdx.x * blockDim.x + threadIdx.x;
  int e    = gt >> 5;
  int lane = gt & 31;
  if (e >= E) return;

  if (e + 8 < E) {                       // global_prefetch_b8 a few edges ahead
    int s2 = src[e + 8];
    __builtin_prefetch(x + (size_t)s2 * H + lane * 4, 0, 0);
  }

  int s = src[e];
  int d = dst[e];
  int et = etype[e];
  et = et < 0 ? 0 : (et > BOND_V - 1 ? BOND_V - 1 : et);

  float4 xv = *(const float4*)(x + (size_t)s * H + lane * 4);
  float4 bv = *(const float4*)(bond_emb + (size_t)et * H + lane * 4);
  float4 m;
  m.x = fmaxf(xv.x + bv.x, 0.f);
  m.y = fmaxf(xv.y + bv.y, 0.f);
  m.z = fmaxf(xv.z + bv.z, 0.f);
  m.w = fmaxf(xv.w + bv.w, 0.f);

  float* ap = agg + (size_t)d * H + lane * 4;
  atomicAdd(ap + 0, m.x);
  atomicAdd(ap + 1, m.y);
  atomicAdd(ap + 2, m.z);
  atomicAdd(ap + 3, m.w);
}

// ---------------------------------------------------------------------------
// TDM: DMA one packed weight matrix (32 KB, flat 1-D tile of 4096 x 8B) into
// LDS. D# built per CDNA5 ISA ch.8: group0 = {count, lds_addr, global_addr,
// type=2}; group1 = {data_size=3, tensor_dim0=tile_dim0=4096, dim1=1,
// stride0=4096}; groups 2/3 zero (<=2-D tensor).
// ---------------------------------------------------------------------------
#if __has_builtin(__builtin_amdgcn_tensor_load_to_lds)
#define USE_TDM 1
__device__ __forceinline__ void tdm_load_weights(const _Float16* gsrc,
                                                 unsigned lds_off) {
  unsigned long long ga = (unsigned long long)(uintptr_t)gsrc;
  u32x4 g0;
  g0.x = 1u;                                        // count=1 (valid, user)
  g0.y = lds_off;                                   // lds_addr (bytes)
  g0.z = (unsigned)(ga & 0xFFFFFFFFu);              // global_addr[31:0]
  g0.w = (unsigned)((ga >> 32) & 0x01FFFFFFu)       // global_addr[56:32]
         | (2u << 30);                              // type = 2 ("image")
  i32x8 g1;
  g1[0] = 0x00030000;   // workgroup_mask=0, data_size=3 (8 B)
  g1[1] = 0x10000000;   // tensor_dim0[15:0]=4096 at bits[63:48]
  g1[2] = 0x00010000;   // tensor_dim1=1 at bits[95:80]
  g1[3] = 0x10000000;   // tile_dim0=4096 at bits[127:112]
  g1[4] = 0x00000001;   // tile_dim1=1
  g1[5] = 4096;         // tensor_dim0_stride=4096 (8B units)
  g1[6] = 0;
  g1[7] = 0;
  i32x4 z4 = {0, 0, 0, 0};
#if defined(__clang_major__) && (__clang_major__ >= 23)
  i32x8 z8 = {0, 0, 0, 0, 0, 0, 0, 0};
  __builtin_amdgcn_tensor_load_to_lds(g0, g1, z4, z4, z8, 0);
#else
  __builtin_amdgcn_tensor_load_to_lds(g0, g1, z4, z4, 0);
#endif
}
#else
#define USE_TDM 0
#endif

// ---------------------------------------------------------------------------
// Fused two-layer MLP via WMMA f16->f32:
//   h = act(in @ W1 + b1);  out = h @ W2 + b2
// ACT: 0 = ReLU (conv layers), 1 = SiLU (projection head)
// Block = 256 threads = 8 waves; each wave owns a 16-row tile, full N=128.
// Both weight matrices DMA'd into LDS once per block via the Tensor Data
// Mover (TENSORcnt); hidden activations staged in per-wave LDS as f16.
// Dynamic LDS: [W1s 32KB][W2s 32KB][hbuf 32KB] = 96 KB.
// ---------------------------------------------------------------------------
template <int ACT>
__global__ void fused_mlp_kernel(const float* __restrict__ in,
                                 const _Float16* __restrict__ w1p,
                                 const float* __restrict__ b1,
                                 const _Float16* __restrict__ w2p,
                                 const float* __restrict__ b2,
                                 float* __restrict__ out,
                                 int rows) {
  extern __shared__ _Float16 smem[];
  _Float16* w1s = smem;                      // 16384 halves
  _Float16* w2s = smem + H * H;              // 16384 halves
  const int wid   = threadIdx.x >> 5;
  const int lane  = threadIdx.x & 31;
  const int half  = lane >> 4;               // 0/1: K-half selector
  const int ln16  = lane & 15;
  const int mbase = (blockIdx.x * 8 + wid) * 16;
  _Float16* hb = smem + 2 * H * H + wid * (16 * H);   // per-wave hidden tile

  // ---- stage both weight matrices into LDS ----
#if USE_TDM
  if (threadIdx.x < 32) {
    tdm_load_weights(w1p, (unsigned)(uintptr_t)w1s);
    tdm_load_weights(w2p, (unsigned)(uintptr_t)w2s);
    __builtin_amdgcn_s_wait_tensorcnt(0);
  }
  __syncthreads();
#else
  {
    const uint4* s1 = (const uint4*)w1p;
    const uint4* s2 = (const uint4*)w2p;
    uint4* d1 = (uint4*)w1s;
    uint4* d2 = (uint4*)w2s;
    for (int i = threadIdx.x; i < (H * H) / 8; i += blockDim.x) {
      d1[i] = s1[i];
      d2[i] = s2[i];
    }
  }
  __syncthreads();
#endif

  // ---- GEMM1: c1[nt] accumulates in f32 over 4 k-steps of 32 ----
  v8f c1[HT];
#pragma unroll
  for (int nt = 0; nt < HT; ++nt)
    c1[nt] = (v8f){0.f, 0.f, 0.f, 0.f, 0.f, 0.f, 0.f, 0.f};

  int mrow = mbase + ln16;
  if (mrow >= rows) mrow = rows - 1;         // clamped loads; stores guarded
  const float* rowp = in + (size_t)mrow * H;

#pragma unroll
  for (int kt = 0; kt < KST; ++kt) {
    // A fragment (16-bit A 16x32 layout): two contiguous 8-float chunks
    const float* rp = rowp + kt * 32 + 8 * half;
    float4 f0 = *(const float4*)(rp + 0);
    float4 f1 = *(const float4*)(rp + 4);
    float4 f2 = *(const float4*)(rp + 16);
    float4 f3 = *(const float4*)(rp + 20);
    v16h a;
    a[0]  = (_Float16)f0.x;  a[1]  = (_Float16)f0.y;
    a[2]  = (_Float16)f0.z;  a[3]  = (_Float16)f0.w;
    a[4]  = (_Float16)f1.x;  a[5]  = (_Float16)f1.y;
    a[6]  = (_Float16)f1.z;  a[7]  = (_Float16)f1.w;
    a[8]  = (_Float16)f2.x;  a[9]  = (_Float16)f2.y;
    a[10] = (_Float16)f2.z;  a[11] = (_Float16)f2.w;
    a[12] = (_Float16)f3.x;  a[13] = (_Float16)f3.y;
    a[14] = (_Float16)f3.z;  a[15] = (_Float16)f3.w;

#pragma unroll
    for (int nt = 0; nt < HT; ++nt) {
      v16h b = *(const v16h*)(w1s + (((kt * HT + nt) * 32 + lane) << 4));
      c1[nt] = __builtin_amdgcn_wmma_f32_16x16x32_f16(
          false, a, false, b, (short)0, c1[nt], false, false);
    }
  }

  // ---- epilogue 1: bias + activation, stage f16 hidden tile in LDS ----
#pragma unroll
  for (int nt = 0; nt < HT; ++nt) {
    int n = nt * 16 + ln16;
    float bias = b1[n];
#pragma unroll
    for (int r = 0; r < 8; ++r) {
      float v = c1[nt][r] + bias;
      if (ACT == 0) v = fmaxf(v, 0.f);
      else          v = v / (1.f + __expf(-v));   // SiLU
      hb[(r + 8 * half) * H + n] = (_Float16)v;
    }
  }

  // ---- GEMM2 from LDS (ds_load_b128 A-fragments) ----
  v8f c2[HT];
#pragma unroll
  for (int nt = 0; nt < HT; ++nt)
    c2[nt] = (v8f){0.f, 0.f, 0.f, 0.f, 0.f, 0.f, 0.f, 0.f};

#pragma unroll
  for (int kt = 0; kt < KST; ++kt) {
    const _Float16* rp2 = hb + ln16 * H + kt * 32 + 8 * half;
    v8h lo = *(const v8h*)(rp2 + 0);
    v8h hi = *(const v8h*)(rp2 + 16);
    v16h a2;
#pragma unroll
    for (int i = 0; i < 8; ++i) { a2[i] = lo[i]; a2[i + 8] = hi[i]; }

#pragma unroll
    for (int nt = 0; nt < HT; ++nt) {
      v16h b = *(const v16h*)(w2s + (((kt * HT + nt) * 32 + lane) << 4));
      c2[nt] = __builtin_amdgcn_wmma_f32_16x16x32_f16(
          false, a2, false, b, (short)0, c2[nt], false, false);
    }
  }

  // ---- epilogue 2: bias + store f32 ----
#pragma unroll
  for (int nt = 0; nt < HT; ++nt) {
    int n = nt * 16 + ln16;
    float bias = b2[n];
#pragma unroll
    for (int r = 0; r < 8; ++r) {
      int m = mbase + r + 8 * half;
      if (m < rows) out[(size_t)m * H + n] = c2[nt][r] + bias;
    }
  }
}

// ---------------------------------------------------------------------------
// Readout: per-graph sum + count (one wave per node, f32 atomics)
// ---------------------------------------------------------------------------
__global__ void readout_sum_kernel(const float* __restrict__ x,
                                   const int* __restrict__ node_batch,
                                   float* __restrict__ gsum,
                                   float* __restrict__ gcnt, int N) {
  int gt   = blockIdx.x * blockDim.x + threadIdx.x;
  int node = gt >> 5;
  int lane = gt & 31;
  if (node >= N) return;
  int g = node_batch[node];
  float4 v = *(const float4*)(x + (size_t)node * H + lane * 4);
  float* sp = gsum + (size_t)g * H + lane * 4;
  atomicAdd(sp + 0, v.x);
  atomicAdd(sp + 1, v.y);
  atomicAdd(sp + 2, v.z);
  atomicAdd(sp + 3, v.w);
  if (lane == 0) atomicAdd(gcnt + g, 1.0f);
}

__global__ void finalize_mean_kernel(const float* __restrict__ gsum,
                                     const float* __restrict__ gcnt,
                                     float* __restrict__ gbuf, int GH) {
  int t = blockIdx.x * blockDim.x + threadIdx.x;
  if (t >= GH) return;
  gbuf[t] = gsum[t] / fmaxf(gcnt[t >> 7], 1.0f);
}

// ---------------------------------------------------------------------------
extern "C" void kernel_launch(void* const* d_in, const int* in_sizes, int n_in,
                              void* d_out, int out_size, void* d_ws, size_t ws_size,
                              hipStream_t stream) {
  const int N = in_sizes[0];
  const int E = in_sizes[2];
  const int G = out_size / H;
  const int L = in_sizes[6] / (H * H);

  const int*   atom_types = (const int*)d_in[0];
  const int*   edges      = (const int*)d_in[1];   // [2,E]: src then dst
  const int*   edge_types = (const int*)d_in[2];
  const int*   node_batch = (const int*)d_in[3];
  const float* atom_emb   = (const float*)d_in[4];
  const float* bond_emb   = (const float*)d_in[5];
  const float* conv_w1    = (const float*)d_in[6];
  const float* conv_b1    = (const float*)d_in[7];
  const float* conv_w2    = (const float*)d_in[8];
  const float* conv_b2    = (const float*)d_in[9];
  const float* proj_w1    = (const float*)d_in[10];
  const float* proj_b1    = (const float*)d_in[11];
  const float* proj_w2    = (const float*)d_in[12];
  const float* proj_b2    = (const float*)d_in[13];

  // ---- workspace carve-up ----
  char* ws = (char*)d_ws;
  float* xbuf = (float*)ws;  ws += (size_t)N * H * sizeof(float);
  float* agg  = (float*)ws;  ws += (size_t)N * H * sizeof(float);
  float* gsum = (float*)ws;  ws += (size_t)G * H * sizeof(float);
  float* gcnt = (float*)ws;  ws += (size_t)G * sizeof(float);
  float* gbuf = (float*)ws;  ws += (size_t)G * H * sizeof(float);
  _Float16* wpack = (_Float16*)ws;  // 2L+2 matrices x H*H halves

  const size_t MAT = (size_t)H * H;               // 16384 elements per matrix
  _Float16* w1p[8];
  _Float16* w2p[8];
  for (int l = 0; l < L; ++l) {
    w1p[l] = wpack + (size_t)(2 * l + 0) * MAT;
    w2p[l] = wpack + (size_t)(2 * l + 1) * MAT;
  }
  _Float16* pw1 = wpack + (size_t)(2 * L + 0) * MAT;
  _Float16* pw2 = wpack + (size_t)(2 * L + 1) * MAT;

  // ---- pack weights to WMMA B-fragment layout (f16) ----
  const int packBlocks = (H * H + 255) / 256;
  for (int l = 0; l < L; ++l) {
    pack_weight_kernel<<<packBlocks, 256, 0, stream>>>(conv_w1 + (size_t)l * MAT, w1p[l]);
    pack_weight_kernel<<<packBlocks, 256, 0, stream>>>(conv_w2 + (size_t)l * MAT, w2p[l]);
  }
  pack_weight_kernel<<<packBlocks, 256, 0, stream>>>(proj_w1, pw1);
  pack_weight_kernel<<<packBlocks, 256, 0, stream>>>(proj_w2, pw2);

  // ---- node embedding gather ----
  {
    long long th = (long long)N * 32;
    embed_nodes_kernel<<<(int)((th + 255) / 256), 256, 0, stream>>>(
        atom_types, atom_emb, xbuf, N);
  }

  const size_t SMEM = (size_t)(3 * H * H) * sizeof(_Float16);   // 96 KB

  // ---- GINE layers ----
  const int mlpBlocksN = (N + 127) / 128;
  for (int l = 0; l < L; ++l) {
    // agg := x  (so scatter produces h = x + sum(messages))
    hipMemcpyAsync(agg, xbuf, (size_t)N * H * sizeof(float),
                   hipMemcpyDeviceToDevice, stream);
    {
      long long th = (long long)E * 32;
      edge_message_kernel<<<(int)((th + 255) / 256), 256, 0, stream>>>(
          xbuf, bond_emb, edges, edges + E, edge_types, agg, E);
    }
    fused_mlp_kernel<0><<<mlpBlocksN, 256, SMEM, stream>>>(
        agg, w1p[l], conv_b1 + (size_t)l * H,
        w2p[l], conv_b2 + (size_t)l * H, xbuf, N);
  }

  // ---- per-graph mean readout ----
  hipMemsetAsync(gsum, 0, (size_t)G * H * sizeof(float), stream);
  hipMemsetAsync(gcnt, 0, (size_t)G * sizeof(float), stream);
  {
    long long th = (long long)N * 32;
    readout_sum_kernel<<<(int)((th + 255) / 256), 256, 0, stream>>>(
        xbuf, node_batch, gsum, gcnt, N);
  }
  {
    int GH = G * H;
    finalize_mean_kernel<<<(GH + 255) / 256, 256, 0, stream>>>(gsum, gcnt, gbuf, GH);
  }

  // ---- projection head: Linear -> SiLU -> Linear ----
  fused_mlp_kernel<1><<<(G + 127) / 128, 256, SMEM, stream>>>(
      gbuf, pw1, proj_b1, pw2, proj_b2, (float*)d_out, G);
}
